// SAFusionBlock_32349693673852
// MI455X (gfx1250) — matrix operand order repaired and checked by reading, not compiled, not run on hardware
//
#include <hip/hip_runtime.h>

// ---------------------------------------------------------------------------
// MI455X (gfx1250) implementation.
//  * ~189 GFLOP total, ~99% in two convs -> v_wmma_f32_16x16x32_f16 implicit GEMM.
//  * Conv tiles: M=128 (8 waves x 16 oc), N=64 pixels (4 B-frags per wave ->
//    4 WMMAs per A-fragment, 4 independent accumulation chains).
//  * Row staging via GLOBAL_LOAD_ASYNC_TO_LDS_B128 (ASYNCcnt) + s_wait_asynccnt.
//  * Channel layout permuted to [F:0..127 | cv:128..176 | pad:177..191] (192=6*32);
//    permutation folded into the weight repack.
// ---------------------------------------------------------------------------

typedef __attribute__((ext_vector_type(16))) _Float16 v16h;
typedef __attribute__((ext_vector_type(8)))  _Float16 v8h;
typedef __attribute__((ext_vector_type(8)))  float    v8f;

#define BN   4
#define CC   128
#define HH   128
#define WW   128
#define CP   192          // padded conv1 input channels (177 -> 192 = 6*32)
#define NPIX (HH*WW)

__device__ __forceinline__ v8f zero8f() {
  v8f z;
#pragma unroll
  for (int i = 0; i < 8; ++i) z[i] = 0.f;
  return z;
}
__device__ __forceinline__ v8h zero8h() {
  v8h z;
#pragma unroll
  for (int i = 0; i < 8; ++i) z[i] = (_Float16)0.f;
  return z;
}

// Async copy 16B global -> LDS (CDNA5 GLOBAL_LOAD_ASYNC_TO_LDS_B128, ASYNCcnt).
// LDS address = low 32 bits of the flat shared pointer (aperture rule: flat
// addr[31:0] is the LDS-relative byte offset).
__device__ __forceinline__ void async_g2l_b128(_Float16* lds_dst, const _Float16* gsrc) {
  unsigned lds = (unsigned)(unsigned long long)(void*)lds_dst;
  unsigned long long ga = (unsigned long long)(const void*)gsrc;
  asm volatile("global_load_async_to_lds_b128 %0, %1, off" :: "v"(lds), "v"(ga) : "memory");
}
__device__ __forceinline__ void wait_async() {
  asm volatile("s_wait_asynccnt 0" ::: "memory");
}

// ---------------------------------------------------------------------------
// 1) NCHW f32 -> NHWC f16 (F channels at 0..127), zero the pad channels.
//    X1 gets Fe, X2 gets Fi.
// ---------------------------------------------------------------------------
__global__ void pack_kernel(const float* __restrict__ Fi,
                            const float* __restrict__ Fe,
                            _Float16* __restrict__ X1,
                            _Float16* __restrict__ X2) {
  int i = blockIdx.x * 256 + threadIdx.x;         // over B*C*H*W = 8388608
  int w = i & 127, h = (i >> 7) & 127, c = (i >> 14) & 127, b = i >> 21;
  float fi = Fi[i], fe = Fe[i];
  long p = ((long)(b * HH + h) * WW + w) * CP;
  X1[p + c] = (_Float16)fe;
  X2[p + c] = (_Float16)fi;
  if (c < CP - 177) {                              // zero pad channels 177..191
    X1[p + 177 + c] = (_Float16)0.f;
    X2[p + 177 + c] = (_Float16)0.f;
  }
}

// ---------------------------------------------------------------------------
// 2) Weight repack (permuted + padded), f32 OIHW -> f16 [ky][kx][oc][cin'].
// ---------------------------------------------------------------------------
__global__ void repack_w1_kernel(const float* __restrict__ w1,
                                 _Float16* __restrict__ w1p) {
  int i = blockIdx.x * 256 + threadIdx.x;         // 5*5*128*192 = 614400
  if (i >= 5 * 5 * 128 * CP) return;
  int cp = i % CP;
  int t  = i / CP;
  int oc = t % 128;
  int k  = t / 128;                                // ky*5+kx
  int ky = k / 5, kx = k % 5;
  float v = 0.f;
  if (cp < 128)      v = w1[((oc * 177 + (49 + cp)) * 5 + ky) * 5 + kx];   // F part
  else if (cp < 177) v = w1[((oc * 177 + (cp - 128)) * 5 + ky) * 5 + kx];  // cv part
  w1p[i] = (_Float16)v;
}

__global__ void repack_w2_kernel(const float* __restrict__ w2,
                                 _Float16* __restrict__ w2p) {
  int i = blockIdx.x * 256 + threadIdx.x;         // 3*3*128*128 = 147456
  if (i >= 3 * 3 * 128 * 128) return;
  int c  = i % 128;
  int t  = i / 128;
  int oc = t % 128;
  int k  = t / 128;
  int ky = k / 3, kx = k % 3;
  w2p[i] = (_Float16)w2[((oc * 128 + c) * 3 + ky) * 3 + kx];
}

// ---------------------------------------------------------------------------
// 3) Cost volume: cv[b,d,h,w] = sum_c center[c]*window(d)[c] / sqrt(128).
//    One thread per (branch,b,h,w). Clamped addrs + final mask (branch-free).
// ---------------------------------------------------------------------------
__global__ void costvol_kernel(_Float16* __restrict__ X1,
                               _Float16* __restrict__ X2) {
  int g = blockIdx.x * 256 + threadIdx.x;         // 2*4*128*128 = 131072
  int w = g & 127, h = (g >> 7) & 127, b = (g >> 14) & 3, br = g >> 16;
  _Float16*       Xc = br ? X2 : X1;
  const _Float16* Xw = br ? X1 : X2;
  long pix = (long)(b * HH + h) * WW + w;
  const _Float16* cen = Xc + pix * CP;

  int   woff[49];
  float msk[49];
#pragma unroll
  for (int dy = 0; dy < 7; ++dy) {
#pragma unroll
    for (int dx = 0; dx < 7; ++dx) {
      int hh = h + dy - 3, ww = w + dx - 3;
      bool ok = (hh >= 0) & (hh < HH) & (ww >= 0) & (ww < WW);
      int hc = hh < 0 ? 0 : (hh > HH - 1 ? HH - 1 : hh);
      int wc = ww < 0 ? 0 : (ww > WW - 1 ? WW - 1 : ww);
      woff[dy * 7 + dx] = ((b * HH + hc) * WW + wc) * CP;
      msk[dy * 7 + dx]  = ok ? 0.08838834764831843f : 0.f;  // 1/sqrt(128) or 0
    }
  }
  float acc[49];
#pragma unroll
  for (int d = 0; d < 49; ++d) acc[d] = 0.f;

  for (int ck = 0; ck < 8; ++ck) {                 // 128 channels / 16
    v16h c16 = *(const v16h*)(cen + ck * 16);
    float cf[16];
#pragma unroll
    for (int i = 0; i < 16; ++i) cf[i] = (float)c16[i];
#pragma unroll
    for (int d = 0; d < 49; ++d) {
      v16h w16 = *(const v16h*)(Xw + woff[d] + ck * 16);
#pragma unroll
      for (int i = 0; i < 16; ++i) acc[d] += cf[i] * (float)w16[i];
    }
  }
  _Float16* outp = Xc + pix * CP + 128;            // cv channels 128..176
#pragma unroll
  for (int d = 0; d < 49; ++d) outp[d] = (_Float16)(acc[d] * msk[d]);
}

// ---------------------------------------------------------------------------
// WMMA fragment helpers (wave32; A/B in this code's contiguous k order, not ISA table order).
//   A 16x32: lane m=lane&15, kb=(lane<16)?0:16; halves h -> K=kb+h
//            (two contiguous 16B runs, 32B total).
//   B 32x16: lane n=lane&15, kb=(lane<16)?0:16; halves h -> K=kb+h (32B run).
//   C/D 16x16 f32: VGPR j -> M = j + 8*(lane>=16), N = lane&15.
// ---------------------------------------------------------------------------
__device__ __forceinline__ v16h load_a_frag(const _Float16* p) {
  v8h a0 = *(const v8h*)(p);
  v8h a1 = *(const v8h*)(p + 8);
  v16h A;
#pragma unroll
  for (int i = 0; i < 8; ++i) { A[i] = a0[i]; A[i + 8] = a1[i]; }
  return A;
}

#define WMMA_F16(A, B, C) \
  __builtin_amdgcn_wmma_f32_16x16x32_f16(false, (A), false, (B), (short)0, (C), false, false)

// ---------------------------------------------------------------------------
// 4) Conv1 (5x5, 192->128) + bias + LeakyReLU(0.1) as implicit GEMM.
//    Block = 8 waves; wave wv owns oc tile [16*wv,16*wv+16); N tile = 64 px.
//    Per ky: async-stage 68px x 192ch row into LDS, then kx(5) x ck(6)
//    K-steps, 4 WMMAs per step sharing one A (weights) fragment.
// ---------------------------------------------------------------------------
#define PXT1 68
__global__ void __launch_bounds__(256)
conv1_wmma_kernel(const _Float16* __restrict__ X1,
                  const _Float16* __restrict__ X2,
                  const _Float16* __restrict__ w1p,
                  const float* __restrict__ b1,
                  _Float16* __restrict__ Y) {
  __shared__ alignas(32) _Float16 row[PXT1 * CP];  // 13056 halves = 25.5KB

  int w0 = blockIdx.x * 64;
  int h  = blockIdx.y;
  int br = blockIdx.z >> 2, b = blockIdx.z & 3;
  const _Float16* X = br ? X2 : X1;

  int tid = threadIdx.x, lane = tid & 31, wv = tid >> 5;
  int oc0 = wv * 16;
  int n   = lane & 15;
  int hi  = lane >> 4;
  int kA  = hi ? 16 : 0;                           // A-fragment K base
  int kB  = hi ? 16 : 0;                           // B-fragment K base

  v8f acc0 = zero8f(), acc1 = zero8f(), acc2 = zero8f(), acc3 = zero8f();

  for (int ky = 0; ky < 5; ++ky) {
    __syncthreads();                               // prev reads done before restage
    int hs = h + ky - 2;
    for (int t = tid; t < PXT1 * (CP / 8); t += 256) {   // 1632 16B units
      int c8 = t % (CP / 8);
      int px = t / (CP / 8);
      int wsx = w0 - 2 + px;
      _Float16* dst = row + px * CP + c8 * 8;
      if (hs >= 0 && hs < HH && wsx >= 0 && wsx < WW)
        async_g2l_b128(dst, X + ((long)(b * HH + hs) * WW + wsx) * CP + c8 * 8);
      else
        *(v8h*)dst = zero8h();                     // zero halo
    }
    wait_async();
    __syncthreads();

    for (int kx = 0; kx < 5; ++kx) {
      const _Float16* wp = w1p + (long)(((ky * 5 + kx) * 128) + oc0 + n) * CP + kA;
      const _Float16* l0 = row + (kx + n) * CP + kB;
      __builtin_prefetch(wp + CP, 0, 1);           // next weight block -> global_prefetch
#pragma unroll
      for (int ck = 0; ck < 6; ++ck) {
        v16h A  = load_a_frag(wp + ck * 32);
        v16h B0 = *(const v16h*)(l0 + 0 * 16 * CP + ck * 32);
        v16h B1 = *(const v16h*)(l0 + 1 * 16 * CP + ck * 32);
        v16h B2 = *(const v16h*)(l0 + 2 * 16 * CP + ck * 32);
        v16h B3 = *(const v16h*)(l0 + 3 * 16 * CP + ck * 32);
        acc0 = WMMA_F16(A, B0, acc0);
        acc1 = WMMA_F16(A, B1, acc1);
        acc2 = WMMA_F16(A, B2, acc2);
        acc3 = WMMA_F16(A, B3, acc3);
      }
    }
  }

  // Epilogue: bias + LeakyReLU(0.1), f16 NHWC store (8 contiguous oc per lane).
  int ocb = oc0 + 8 * hi;
  v8f accs[4] = {acc0, acc1, acc2, acc3};
  long pixbase = (long)br * (BN * NPIX) + (long)(b * HH + h) * WW;
#pragma unroll
  for (int f = 0; f < 4; ++f) {
    v8h o;
#pragma unroll
    for (int j = 0; j < 8; ++j) {
      float v = accs[f][j] + b1[ocb + j];
      v = v > 0.f ? v : 0.1f * v;
      o[j] = (_Float16)v;
    }
    *(v8h*)(Y + (pixbase + w0 + 16 * f + n) * CC + ocb) = o;
  }
}

// ---------------------------------------------------------------------------
// 5) Conv2 (3x3, 128->128) + bias, f32 NCHW output straight into d_out.
// ---------------------------------------------------------------------------
#define PXT2 66
__global__ void __launch_bounds__(256)
conv2_wmma_kernel(const _Float16* __restrict__ Y,
                  const _Float16* __restrict__ w2p,
                  const float* __restrict__ b2,
                  float* __restrict__ out) {
  __shared__ alignas(32) _Float16 row[PXT2 * CC];  // 8448 halves = 16.5KB

  int w0 = blockIdx.x * 64;
  int h  = blockIdx.y;
  int br = blockIdx.z >> 2, b = blockIdx.z & 3;

  int tid = threadIdx.x, lane = tid & 31, wv = tid >> 5;
  int oc0 = wv * 16;
  int n   = lane & 15;
  int hi  = lane >> 4;
  int kA  = hi ? 16 : 0;
  int kB  = hi ? 16 : 0;

  v8f acc0 = zero8f(), acc1 = zero8f(), acc2 = zero8f(), acc3 = zero8f();

  for (int ky = 0; ky < 3; ++ky) {
    __syncthreads();
    int hs = h + ky - 1;
    for (int t = tid; t < PXT2 * (CC / 8); t += 256) {   // 1056 16B units
      int c8 = t % (CC / 8);
      int px = t / (CC / 8);
      int wsx = w0 - 1 + px;
      _Float16* dst = row + px * CC + c8 * 8;
      if (hs >= 0 && hs < HH && wsx >= 0 && wsx < WW)
        async_g2l_b128(dst, Y + ((long)br * (BN * NPIX) +
                                 (long)(b * HH + hs) * WW + wsx) * CC + c8 * 8);
      else
        *(v8h*)dst = zero8h();
    }
    wait_async();
    __syncthreads();

    for (int kx = 0; kx < 3; ++kx) {
      const _Float16* wp = w2p + (long)(((ky * 3 + kx) * 128) + oc0 + n) * CC + kA;
      const _Float16* l0 = row + (kx + n) * CC + kB;
      __builtin_prefetch(wp + CC, 0, 1);
#pragma unroll
      for (int ck = 0; ck < 4; ++ck) {
        v16h A  = load_a_frag(wp + ck * 32);
        v16h B0 = *(const v16h*)(l0 + 0 * 16 * CC + ck * 32);
        v16h B1 = *(const v16h*)(l0 + 1 * 16 * CC + ck * 32);
        v16h B2 = *(const v16h*)(l0 + 2 * 16 * CC + ck * 32);
        v16h B3 = *(const v16h*)(l0 + 3 * 16 * CC + ck * 32);
        acc0 = WMMA_F16(A, B0, acc0);
        acc1 = WMMA_F16(A, B1, acc1);
        acc2 = WMMA_F16(A, B2, acc2);
        acc3 = WMMA_F16(A, B3, acc3);
      }
    }
  }

  // Epilogue: bias, f32 NCHW scatter into d_out (branch-concatenated).
  int ocb = oc0 + 8 * hi;
  v8f accs[4] = {acc0, acc1, acc2, acc3};
  long obase = (long)br * (BN * CC * NPIX) + (long)b * (CC * NPIX) + (long)h * WW;
#pragma unroll
  for (int f = 0; f < 4; ++f) {
#pragma unroll
    for (int j = 0; j < 8; ++j) {
      out[obase + (long)(ocb + j) * NPIX + (w0 + 16 * f + n)] = accs[f][j] + b2[ocb + j];
    }
  }
}

// ---------------------------------------------------------------------------
// Launch: pack -> weight repack -> cost volume -> conv1(WMMA) -> conv2(WMMA).
// Workspace (bytes): X1 24M | X2 24M | Y 32M | w1p 1.2M | w2p 0.29M  (~82MB)
// ---------------------------------------------------------------------------
extern "C" void kernel_launch(void* const* d_in, const int* in_sizes, int n_in,
                              void* d_out, int out_size, void* d_ws, size_t ws_size,
                              hipStream_t stream) {
  (void)in_sizes; (void)n_in; (void)out_size; (void)ws_size;
  const float* Fi = (const float*)d_in[0];
  const float* Fe = (const float*)d_in[1];
  const float* w1 = (const float*)d_in[2];
  const float* b1 = (const float*)d_in[3];
  const float* w2 = (const float*)d_in[4];
  const float* b2 = (const float*)d_in[5];
  float* out = (float*)d_out;

  char* ws = (char*)d_ws;
  const size_t XBYTES = (size_t)BN * NPIX * CP * 2;      // 25165824
  const size_t YBYTES = (size_t)2 * BN * NPIX * CC * 2;  // 33554432
  _Float16* X1  = (_Float16*)(ws);
  _Float16* X2  = (_Float16*)(ws + XBYTES);
  _Float16* Y   = (_Float16*)(ws + 2 * XBYTES);
  _Float16* w1p = (_Float16*)(ws + 2 * XBYTES + YBYTES);
  _Float16* w2p = (_Float16*)(ws + 2 * XBYTES + YBYTES + (size_t)5*5*128*CP*2);

  pack_kernel<<<(BN * CC * NPIX) / 256, 256, 0, stream>>>(Fi, Fe, X1, X2);
  repack_w1_kernel<<<(5 * 5 * 128 * CP) / 256, 256, 0, stream>>>(w1, w1p);
  repack_w2_kernel<<<(3 * 3 * 128 * 128) / 256, 256, 0, stream>>>(w2, w2p);
  costvol_kernel<<<(2 * BN * NPIX) / 256, 256, 0, stream>>>(X1, X2);

  dim3 cgrid(WW / 64, HH, 2 * BN);
  conv1_wmma_kernel<<<cgrid, 256, 0, stream>>>(X1, X2, w1p, b1, Y);
  conv2_wmma_kernel<<<cgrid, 256, 0, stream>>>(Y, w2p, b2, out);
}
